// SATCausalAttention_23467701305699
// MI455X (gfx1250) — matrix operand order repaired and checked by reading, not compiled, not run on hardware
//
#include <hip/hip_runtime.h>
#include <hip/hip_fp16.h>

typedef __attribute__((ext_vector_type(16))) _Float16 v16h;
typedef __attribute__((ext_vector_type(8)))  _Float16 v8h;
typedef __attribute__((ext_vector_type(8)))  float    v8f;
typedef __attribute__((ext_vector_type(4)))  unsigned int v4u;
typedef __attribute__((ext_vector_type(8)))  int v8i;
typedef __attribute__((ext_vector_type(4)))  int v4i;

constexpr float SCALE = 0.125f;   // 1/sqrt(64)
constexpr float NEGM  = -1e9f;

#if __has_builtin(__builtin_amdgcn_tensor_load_to_lds)
#define HAVE_TDM 1
#else
#define HAVE_TDM 0
#endif

__device__ inline v8f wmma_f16(v16h a, v16h b, v8f c) {
  // v_wmma_f32_16x16x32_f16: (neg_a, A, neg_b, B, c_mod, C, reuse_a, reuse_b)
  return __builtin_amdgcn_wmma_f32_16x16x32_f16(false, a, false, b, (short)0, c, false, false);
}

// A-operand (16x32 f16, MxK): lane = row (lane&15); halves K = base..base+7 and base+16..base+23,
// base = (lane>=16)*8.  rowp points at row start (+k0).
__device__ inline v16h load_a(const _Float16* rowp, int half) {
  union { v16h v; v8h h[2]; } u;
  u.h[0] = *(const v8h*)(rowp + half * 8);
  u.h[1] = *(const v8h*)(rowp + half * 8 + 16);
  return u.v;
}
// B-operand (32x16 f16, KxN) from N-major storage: lane = column (lane&15);
// halves K = base..base+7 and base+16..base+23, base = half*8.  rowp points at column-row start (+k0).
__device__ inline v16h load_b(const _Float16* rowp, int half) {
  union { v16h v; v8h h[2]; } u;
  u.h[0] = *(const v8h*)(rowp + half * 8);
  u.h[1] = *(const v8h*)(rowp + half * 8 + 16);
  return u.v;
}

template <int N>
__device__ inline void wait_tensorcnt() {
#if __has_builtin(__builtin_amdgcn_s_wait_tensorcnt)
  __builtin_amdgcn_s_wait_tensorcnt(N);
#else
  asm volatile("s_wait_tensorcnt %0" :: "i"(N) : "memory");
#endif
}

// Tensor Data Mover: DMA a 2D tile (tile_d0 x tile_d1, f16 elements, row stride stride0)
// from global to LDS.  D# layout per CDNA5 ISA.
__device__ inline void tdm_load_2d(const _Float16* gsrc, _Float16* ldst,
                                   unsigned tensor_d0, unsigned tensor_d1,
                                   unsigned tile_d0, unsigned tile_d1,
                                   unsigned stride0) {
#if HAVE_TDM
  unsigned lds_off = (unsigned)(size_t)(void*)ldst;   // flat-LDS addr[31:0] = LDS byte offset
  unsigned long long ga = (unsigned long long)gsrc;
  v4u g0;
  g0.x = 1u;                                          // count = 1 valid descriptor
  g0.y = lds_off;                                     // lds_addr (bits 63:32)
  g0.z = (unsigned)(ga & 0xFFFFFFFFu);                // global_addr low
  g0.w = (unsigned)((ga >> 32) & 0x1FFFFFFu) | (2u << 30); // global_addr hi | type=2(image)
  v8i g1;
  g1[0] = (int)(1u << 16);                            // workgroup_mask=0, data_size=1 (2B)
  g1[1] = (int)((tensor_d0 & 0xFFFFu) << 16);         // tensor_dim0[15:0] at bits 63:48
  g1[2] = (int)(((tensor_d0 >> 16) & 0xFFFFu) | ((tensor_d1 & 0xFFFFu) << 16));
  g1[3] = (int)(((tensor_d1 >> 16) & 0xFFFFu) | ((tile_d0 & 0xFFFFu) << 16)); // tile_dim0
  g1[4] = (int)(tile_d1 & 0xFFFFu);                   // tile_dim1 (tile_dim2 = 0)
  g1[5] = (int)stride0;                               // tensor_dim0_stride[31:0]
  g1[6] = 0;                                          // stride0 hi / stride1 lo
  g1[7] = 0;
  v4i z4 = {0, 0, 0, 0};
#if __clang_major__ >= 23
  v8i z8 = {0, 0, 0, 0, 0, 0, 0, 0};
  __builtin_amdgcn_tensor_load_to_lds(g0, g1, z4, z4, z8, 0);
#else
  __builtin_amdgcn_tensor_load_to_lds(g0, g1, z4, z4, 0);
#endif
#else
  (void)gsrc; (void)ldst; (void)tensor_d0; (void)tensor_d1;
  (void)tile_d0; (void)tile_d1; (void)stride0;
#endif
}

// ---------------------------------------------------------------- converts
__global__ void cvt_f16(const float* __restrict__ in, _Float16* __restrict__ out, int n) {
  int i = blockIdx.x * 256 + threadIdx.x;
  if (i < n) out[i] = (_Float16)in[i];
}

// out[n][k] = (f16) in[k][n], 1024x1024
__global__ __launch_bounds__(256) void transpose_cvt(const float* __restrict__ in,
                                                     _Float16* __restrict__ out) {
  __shared__ float t[32][33];
  int bx = blockIdx.x * 32, by = blockIdx.y * 32;
  int tx = threadIdx.x & 31, ty = threadIdx.x >> 5;
#pragma unroll
  for (int r = 0; r < 32; r += 8)
    t[ty + r][tx] = in[(size_t)(by + ty + r) * 1024 + bx + tx];
  __syncthreads();
#pragma unroll
  for (int r = 0; r < 32; r += 8)
    out[(size_t)(bx + ty + r) * 1024 + by + tx] = (_Float16)t[tx][ty + r];
}

// ---------------------------------------------------------------- GEMM 4096x1024x1024
// C = A(f16, row-major MxK) * Bt(f16, N-major NxK) + bias
// 64x64 C tile per block (4 waves, 16 rows each), BLK_K = 64, double-buffered TDM pipeline.
// mode 0: C f16 row-major; mode 1: C f16 scattered as Vt[b,h,dh,l]; mode 2: C f32 row-major
__global__ __launch_bounds__(128) void gemm64(const _Float16* __restrict__ A,
                                              const _Float16* __restrict__ Bt,
                                              const float* __restrict__ bias,
                                              void* __restrict__ C, int mode) {
  __shared__ _Float16 As[2][64 * 64];
  __shared__ _Float16 Bs[2][64 * 64];
  const int tid = threadIdx.x, wid = tid >> 5, lane = tid & 31;
  const int m0 = blockIdx.x * 64, n0 = blockIdx.y * 64;
  const int row = lane & 15, half = lane >> 4;
  v8f acc[4] = {v8f{}, v8f{}, v8f{}, v8f{}};

#if HAVE_TDM
  // ---- software-pipelined Tensor Data Mover double buffer ----
  if (wid == 0) {
    tdm_load_2d(A  + (size_t)m0 * 1024, As[0], 1024, 4096, 64, 64, 1024);
    tdm_load_2d(Bt + (size_t)n0 * 1024, Bs[0], 1024, 1024, 64, 64, 1024);
  }
  for (int it = 0; it < 16; ++it) {
    const int cur = it & 1;
    if (wid == 0) {
      if (it + 1 < 16) {
        const int kn = (it + 1) * 64;
        tdm_load_2d(A  + (size_t)m0 * 1024 + kn, As[cur ^ 1], 1024, 4096, 64, 64, 1024);
        tdm_load_2d(Bt + (size_t)n0 * 1024 + kn, Bs[cur ^ 1], 1024, 1024, 64, 64, 1024);
        wait_tensorcnt<2>();   // 2 newest still in flight; current buffer's loads complete
      } else {
        wait_tensorcnt<0>();
      }
    }
    __syncthreads();
    const _Float16* as = As[cur];
    const _Float16* bs = Bs[cur];
#pragma unroll
    for (int ks = 0; ks < 2; ks++) {
      v16h a   = load_a(as + (wid * 16 + row) * 64 + ks * 32, half);
      v16h bm0 = load_b(bs + (0 * 16 + row) * 64 + ks * 32, half);
      v16h bm1 = load_b(bs + (1 * 16 + row) * 64 + ks * 32, half);
      v16h bm2 = load_b(bs + (2 * 16 + row) * 64 + ks * 32, half);
      v16h bm3 = load_b(bs + (3 * 16 + row) * 64 + ks * 32, half);
      acc[0] = wmma_f16(a, bm0, acc[0]);
      acc[1] = wmma_f16(a, bm1, acc[1]);
      acc[2] = wmma_f16(a, bm2, acc[2]);
      acc[3] = wmma_f16(a, bm3, acc[3]);
    }
    __syncthreads();
  }
#else
  // ---- fallback: manual staged copies, single buffer ----
  for (int k0 = 0; k0 < 1024; k0 += 64) {
    const int lr = tid >> 1, lc = (tid & 1) * 32;
#pragma unroll
    for (int q = 0; q < 4; q++) {
      *(v8h*)(As[0] + lr * 64 + lc + q * 8) =
          *(const v8h*)(A + (size_t)(m0 + lr) * 1024 + k0 + lc + q * 8);
      *(v8h*)(Bs[0] + lr * 64 + lc + q * 8) =
          *(const v8h*)(Bt + (size_t)(n0 + lr) * 1024 + k0 + lc + q * 8);
    }
    if (k0 + 64 < 1024) {
      __builtin_prefetch(A + (size_t)(m0 + lr) * 1024 + k0 + 64 + lc, 0, 1);
      __builtin_prefetch(Bt + (size_t)(n0 + lr) * 1024 + k0 + 64 + lc, 0, 1);
    }
    __syncthreads();
#pragma unroll
    for (int ks = 0; ks < 2; ks++) {
      v16h a   = load_a(As[0] + (wid * 16 + row) * 64 + ks * 32, half);
      v16h bm0 = load_b(Bs[0] + (0 * 16 + row) * 64 + ks * 32, half);
      v16h bm1 = load_b(Bs[0] + (1 * 16 + row) * 64 + ks * 32, half);
      v16h bm2 = load_b(Bs[0] + (2 * 16 + row) * 64 + ks * 32, half);
      v16h bm3 = load_b(Bs[0] + (3 * 16 + row) * 64 + ks * 32, half);
      acc[0] = wmma_f16(a, bm0, acc[0]);
      acc[1] = wmma_f16(a, bm1, acc[1]);
      acc[2] = wmma_f16(a, bm2, acc[2]);
      acc[3] = wmma_f16(a, bm3, acc[3]);
    }
    __syncthreads();
  }
#endif

  const int mb = half * 8;
#pragma unroll
  for (int s = 0; s < 4; s++) {
    int n = n0 + s * 16 + row;
    float bn_ = bias[n];
#pragma unroll
    for (int r = 0; r < 8; r++) {
      int m = m0 + wid * 16 + mb + r;
      float v = acc[s][r] + bn_;
      if (mode == 0) {
        ((_Float16*)C)[(size_t)m * 1024 + n] = (_Float16)v;
      } else if (mode == 1) {                       // Vt[b, n, l] with n = h*64+dh
        int bb = m >> 11, l = m & 2047;
        ((_Float16*)C)[((size_t)(bb * 1024 + n)) * 2048 + l] = (_Float16)v;
      } else {
        ((float*)C)[(size_t)m * 1024 + n] = v;
      }
    }
  }
}

// ---------------------------------------------------------------- fused attention
// One block per (b, h, 16-query tile). 128 threads = 4 waves.
__global__ __launch_bounds__(128) void attn_fused(
    const _Float16* __restrict__ Q, const _Float16* __restrict__ K,
    const _Float16* __restrict__ Vt, const float* __restrict__ Bg,
    const float* __restrict__ lattice, const float* __restrict__ Wl,
    const float* __restrict__ bl, float* __restrict__ attn,
    _Float16* __restrict__ Oh) {
  __shared__ float sc[16 * 2048];     // 128 KB score row-panel (fits: 320 KB LDS / WGP)
  __shared__ float red[16 * 8];
  __shared__ float rowstat[16];

  const int blk = blockIdx.x;
  const int qt = blk & 127, h = (blk >> 7) & 15, b = blk >> 11;
  const int tid = threadIdx.x, wid = tid >> 5, lane = tid & 31;
  const int row = lane & 15, half = lane >> 4;
  const int i0 = qt * 16, nkt = qt + 1, ncols = nkt * 16;

  float lb = bl[h];
#pragma unroll
  for (int c = 0; c < 6; c++) lb += lattice[b * 6 + c] * Wl[c * 16 + h];

  // Q A-operand for dh = 0..63 (two K=32 steps), same for every wave
  const _Float16* Qrow = Q + ((size_t)(b * 2048 + i0 + row)) * 1024 + h * 64;
  v16h a0 = load_a(Qrow, half);
  v16h a1 = load_a(Qrow + 32, half);
  const _Float16* Kbase = K + ((size_t)(b * 2048)) * 1024 + h * 64;

  // ---- scores: waves stripe over causal key tiles
  for (int kt = wid; kt < nkt; kt += 4) {
    const _Float16* Krow = Kbase + (size_t)(kt * 16 + row) * 1024;
    v16h b0 = load_b(Krow, half);
    v16h b1 = load_b(Krow + 32, half);
    v8f acc = {};
    acc = wmma_f16(a0, b0, acc);
    acc = wmma_f16(a1, b1, acc);
    int j = kt * 16 + row;                       // lane column = key
#pragma unroll
    for (int r = 0; r < 8; r++) {
      int m = half * 8 + r;                      // query row in tile
      int i = i0 + m;
      float s = acc[r] * SCALE + lb + Bg[((size_t)(b * 2048 + i)) * 2048 + j];
      if (j > i) s += NEGM;
      sc[m * 2048 + j] = s;
    }
  }
  __syncthreads();

  // ---- softmax over the row-panel: 8 threads per row
  const int r16 = tid & 15, sub = tid >> 4;
  float* srow = sc + r16 * 2048;
  float mx = -3.4e38f;
  for (int c = sub; c < ncols; c += 8) mx = fmaxf(mx, srow[c]);
  red[r16 * 8 + sub] = mx;
  __syncthreads();
  if (tid < 16) {
    float m2 = red[tid * 8];
    for (int s = 1; s < 8; s++) m2 = fmaxf(m2, red[tid * 8 + s]);
    rowstat[tid] = m2;
  }
  __syncthreads();
  float rmax = rowstat[r16];
  float sum = 0.f;
  for (int c = sub; c < ncols; c += 8) {
    float e = __expf(srow[c] - rmax);
    srow[c] = e;
    sum += e;
  }
  red[r16 * 8 + sub] = sum;
  __syncthreads();
  if (tid < 16) {
    float s2 = 0.f;
    for (int s = 0; s < 8; s++) s2 += red[tid * 8 + s];
    rowstat[tid] = s2;
  }
  __syncthreads();
  float inv = 1.f / rowstat[r16];
  float* arow = attn + ((size_t)((b * 16 + h) * 2048 + i0 + r16)) * 2048;
  for (int c = sub; c < ncols; c += 8) {
    float p = srow[c] * inv;
    srow[c] = p;
    arow[c] = p;                                  // normalized attn -> d_out
  }
  for (int c = ncols + sub; c < 2048; c += 8) arow[c] = 0.f;  // masked tail
  int ncols32 = (ncols + 31) & ~31;
  if (sub == 0)
    for (int c = ncols; c < ncols32; c++) srow[c] = 0.f;      // pad LDS for PV
  __syncthreads();

  // ---- PV: each wave owns one 16-wide dh tile; A = probs (f32 LDS -> f16), B = Vt
  const int nk2 = (nkt + 1) >> 1;
  v8f oacc = {};
  const _Float16* Vcol = Vt + ((size_t)(b * 1024 + h * 64 + wid * 16 + row)) * 2048;
  for (int k2 = 0; k2 < nk2; k2++) {
    int k0 = k2 * 32;
    union { v16h v; _Float16 hh[16]; } ua;
    const float* sr = sc + row * 2048 + k0;
    int base = half * 8;
#pragma unroll
    for (int t = 0; t < 8; t++) {
      ua.hh[t]     = (_Float16)sr[base + t];
      ua.hh[8 + t] = (_Float16)sr[base + 16 + t];
    }
    v16h bmat = load_b(Vcol + k0, half);
    oacc = wmma_f16(ua.v, bmat, oacc);
  }
#pragma unroll
  for (int r = 0; r < 8; r++) {
    int m = half * 8 + r;
    Oh[((size_t)(b * 2048 + i0 + m)) * 1024 + h * 64 + wid * 16 + row] = (_Float16)oacc[r];
  }
}

// ---------------------------------------------------------------- launch
extern "C" void kernel_launch(void* const* d_in, const int* in_sizes, int n_in,
                              void* d_out, int out_size, void* d_ws, size_t ws_size,
                              hipStream_t stream) {
  const float* x       = (const float*)d_in[0];
  const float* lattice = (const float*)d_in[1];
  const float* Bg      = (const float*)d_in[2];
  // d_in[3] = attn_mask (causal, recomputed on device)
  const float* Wq = (const float*)d_in[4];
  const float* bq = (const float*)d_in[5];
  const float* Wk = (const float*)d_in[6];
  const float* bk = (const float*)d_in[7];
  const float* Wv = (const float*)d_in[8];
  const float* bv = (const float*)d_in[9];
  const float* Wo = (const float*)d_in[10];
  const float* bo = (const float*)d_in[11];
  const float* Wl = (const float*)d_in[12];
  const float* bl = (const float*)d_in[13];

  float* out  = (float*)d_out;
  float* attn = out + (size_t)2 * 2048 * 1024;     // outputs: [out | attn]

  _Float16* ws = (_Float16*)d_ws;
  const size_t MN = (size_t)4096 * 1024;           // 4M halfs
  _Float16* xh  = ws;                              // x in f16
  _Float16* WqT = ws + MN;                         // 4 transposed weights = MN halfs total
  _Float16* WkT = WqT + 1048576;
  _Float16* WvT = WkT + 1048576;
  _Float16* WoT = WvT + 1048576;
  _Float16* Qh  = ws + 2 * MN;                     // [b,l,h,dh] f16
  _Float16* Kh  = ws + 3 * MN;                     // [b,l,h,dh] f16
  _Float16* Vt  = ws + 4 * MN;                     // [b,h,dh,l] f16
  _Float16* Oh  = ws + 5 * MN;                     // [b,l,h*dh] f16

  cvt_f16<<<(int)(MN / 256), 256, 0, stream>>>(x, xh, (int)MN);
  dim3 tg(32, 32);
  transpose_cvt<<<tg, 256, 0, stream>>>(Wq, WqT);
  transpose_cvt<<<tg, 256, 0, stream>>>(Wk, WkT);
  transpose_cvt<<<tg, 256, 0, stream>>>(Wv, WvT);
  transpose_cvt<<<tg, 256, 0, stream>>>(Wo, WoT);

  dim3 gg(64, 16);
  gemm64<<<gg, 128, 0, stream>>>(xh, WqT, bq, Qh, 0);
  gemm64<<<gg, 128, 0, stream>>>(xh, WkT, bk, Kh, 0);
  gemm64<<<gg, 128, 0, stream>>>(xh, WvT, bv, Vt, 1);

  attn_fused<<<4096, 128, 0, stream>>>(Qh, Kh, Vt, Bg, lattice, Wl, bl, attn, Oh);

  gemm64<<<gg, 128, 0, stream>>>(Oh, WoT, bo, out, 2);
}
